// Attention_52896817217709
// MI455X (gfx1250) — matrix operand order repaired and checked by reading, not compiled, not run on hardware
//
#include <hip/hip_runtime.h>
#include <hip/hip_bf16.h>

// Problem constants (match reference)
#define NB 2
#define SS 2048
#define NH 12
#define DMODEL 768
#define DHEAD 64
#define IGNORE_VAL (-100000.0f)

typedef __attribute__((ext_vector_type(16))) __bf16 v16bf;
typedef __attribute__((ext_vector_type(8)))  __bf16 v8bf;
typedef __attribute__((ext_vector_type(8)))  float  v8f;

union BF16x16 { v16bf v; v8bf h[2]; };

__device__ __forceinline__ v8f zero8() {
    v8f z;
#pragma unroll
    for (int i = 0; i < 8; ++i) z[i] = 0.0f;
    return z;
}

// ---- WMMA fragment loaders (layouts per CDNA5 ISA 7.12.2, wave32) ----

// A fragment, 16x32 bf16, row-major source with leading dim `ld`.
// lane l: row = l&15; lanes 0-15 hold K 0..7 & 16..23, lanes 16-31 hold K 8..15 & 24..31.
__device__ __forceinline__ v16bf load_a_bf16(const __bf16* base, int ld, int lane) {
    int r = lane & 15, hl = lane >> 4;
    const __bf16* p = base + (size_t)r * ld + hl * 8;
    BF16x16 f;
    f.h[0] = *(const v8bf*)(p);
    f.h[1] = *(const v8bf*)(p + 16);
    return f.v;
}

// A fragment from fp32 source, converted to bf16 in-register.
__device__ __forceinline__ v16bf load_a_f32(const float* base, int ld, int lane) {
    int r = lane & 15, hl = lane >> 4;
    const float* p = base + (size_t)r * ld + hl * 8;
    v8f a0 = *(const v8f*)(p);
    v8f a1 = *(const v8f*)(p + 16);
    BF16x16 f;
#pragma unroll
    for (int i = 0; i < 8; ++i) { f.v[i] = (__bf16)a0[i]; f.v[8 + i] = (__bf16)a1[i]; }
    return f.v;
}

// B fragment, 32x16 bf16, from B-transposed storage (N rows x K cols, leading dim ld).
// lane l: column n = l&15; lanes 0-15 hold K 0..7 & 16..23, lanes 16-31 hold K 8..15 & 24..31.
__device__ __forceinline__ v16bf load_bt_bf16(const __bf16* base, int ld, int lane) {
    int n = lane & 15, hl = lane >> 4;
    const __bf16* p = base + (size_t)n * ld + hl * 8;
    BF16x16 f;
    f.h[0] = *(const v8bf*)(p);
    f.h[1] = *(const v8bf*)(p + 16);
    return f.v;
}

// ---- Kernel 0: convert + transpose weights to bf16 ----
// Wt  (3, NH, DHEAD, DMODEL) <- W_{Q,K,V} (NH, DMODEL, DHEAD)
// Wot (NH, DMODEL, DHEAD)    <- W_O       (NH, DHEAD, DMODEL)
__global__ void prep_weights(const float* __restrict__ Wq, const float* __restrict__ Wk,
                             const float* __restrict__ Wv, const float* __restrict__ Wo,
                             __bf16* __restrict__ Wt, __bf16* __restrict__ Wot) {
    int idx = blockIdx.x * blockDim.x + threadIdx.x;
    const int per = NH * DMODEL * DHEAD;
    if (idx >= 4 * per) return;
    int t = idx / per;
    int r = idx - t * per;
    int h = r / (DMODEL * DHEAD);
    int r2 = r - h * (DMODEL * DHEAD);
    int mm = r2 / DHEAD;
    int d = r2 - mm * DHEAD;
    if (t < 3) {
        const float* W = (t == 0) ? Wq : ((t == 1) ? Wk : Wv);
        // dst: Wt[t][h][d][mm]
        Wt[(((size_t)t * NH + h) * DHEAD + d) * DMODEL + mm] =
            (__bf16)W[((size_t)h * DMODEL + mm) * DHEAD + d];
    } else {
        // dst: Wot[h][mm][d] = Wo[h][d][mm]
        Wot[((size_t)h * DMODEL + mm) * DHEAD + d] =
            (__bf16)Wo[((size_t)h * DHEAD + d) * DMODEL + mm];
    }
}

// ---- Kernel 1: QKV projection ----
// grid: (SS/64, 3*NH, NB), block 128 (4 waves). Wave computes 16 rows x 64 cols.
__global__ void qkv_proj(const float* __restrict__ Xq, const float* __restrict__ Xk,
                         const float* __restrict__ Xv, const __bf16* __restrict__ Wt,
                         const float* __restrict__ bQ, const float* __restrict__ bK,
                         const float* __restrict__ bV,
                         __bf16* __restrict__ Qb, __bf16* __restrict__ Kb,
                         __bf16* __restrict__ Vt) {
    int wv = threadIdx.x >> 5;
    int lane = threadIdx.x & 31;
    int r16 = lane & 15, hl = lane >> 4;
    int proj = blockIdx.y / NH;
    int h = blockIdx.y % NH;
    int b = blockIdx.z;
    int s0 = blockIdx.x * 64 + wv * 16;

    const float* X = (proj == 0) ? Xq : ((proj == 1) ? Xk : Xv);
    const float* bias = (proj == 0) ? bQ : ((proj == 1) ? bK : bV);
    const __bf16* W = Wt + ((size_t)proj * NH + h) * DHEAD * DMODEL;

    const float* Abase = X + (((size_t)b * SS + s0) * NH + h) * DMODEL;
    const int lda = NH * DMODEL;

    v8f acc[4];
#pragma unroll
    for (int nt = 0; nt < 4; ++nt) acc[nt] = zero8();

    for (int k0 = 0; k0 < DMODEL; k0 += 32) {
        v16bf a = load_a_f32(Abase + k0, lda, lane);
#pragma unroll
        for (int nt = 0; nt < 4; ++nt) {
            v16bf bfrag = load_bt_bf16(W + (size_t)(nt * 16) * DMODEL + k0, DMODEL, lane);
            acc[nt] = __builtin_amdgcn_wmma_f32_16x16x32_bf16(
                false, a, false, bfrag, (short)0, acc[nt], false, false);
        }
    }

    // Epilogue: bias, convert, store. Row = s0 + hl*8 + r, col = nt*16 + r16.
#pragma unroll
    for (int nt = 0; nt < 4; ++nt) {
        int col = nt * 16 + r16;
        float bv = bias[h * DHEAD + col];
#pragma unroll
        for (int r = 0; r < 8; ++r) {
            int row = s0 + hl * 8 + r;
            float val = acc[nt][r] + bv;
            if (proj == 0) {
                Qb[(((size_t)b * NH + h) * SS + row) * DHEAD + col] = (__bf16)val;
            } else if (proj == 1) {
                Kb[(((size_t)b * NH + h) * SS + row) * DHEAD + col] = (__bf16)val;
            } else {
                // V stored transposed: (B, NH, DHEAD, SS)
                Vt[(((size_t)b * NH + h) * DHEAD + col) * SS + row] = (__bf16)val;
            }
        }
    }
}

// ---- Kernel 2: causal flash attention ----
// grid: (SS/64, NH, NB), block 128 (4 waves). Wave handles 16 queries, full key loop.
__global__ void attn(const __bf16* __restrict__ Qb, const __bf16* __restrict__ Kb,
                     const __bf16* __restrict__ Vt, __bf16* __restrict__ Zb) {
    __shared__ __bf16 pbuf[4][16 * 32];

    int wv = threadIdx.x >> 5;
    int lane = threadIdx.x & 31;
    int r16 = lane & 15, hl = lane >> 4;
    int b = blockIdx.z, h = blockIdx.y;
    int q0 = (blockIdx.x * 4 + wv) * 16;

    const __bf16* Q = Qb + ((size_t)b * NH + h) * SS * DHEAD;
    const __bf16* K = Kb + ((size_t)b * NH + h) * SS * DHEAD;
    const __bf16* V = Vt + ((size_t)b * NH + h) * DHEAD * SS;

    // Q fragments for d_head 0..31 and 32..63 (loaded once).
    v16bf qa[2];
#pragma unroll
    for (int kk = 0; kk < 2; ++kk)
        qa[kk] = load_a_bf16(Q + (size_t)q0 * DHEAD + kk * 32, DHEAD, lane);

    v8f o[4];
#pragma unroll
    for (int nt = 0; nt < 4; ++nt) o[nt] = zero8();
    float m[8], lsum[8];
#pragma unroll
    for (int r = 0; r < 8; ++r) { m[r] = -3.0e38f; lsum[r] = 0.0f; }

    const int kend = q0 + 16;  // causal: keys 0..q0+15
    for (int kc = 0; kc < kend; kc += 32) {
        // scores: two 16x16 sub-tiles (keys kc..kc+15, kc+16..kc+31)
        v8f s[2];
#pragma unroll
        for (int sub = 0; sub < 2; ++sub) {
            v8f c = zero8();
#pragma unroll
            for (int kk = 0; kk < 2; ++kk) {
                v16bf kf = load_bt_bf16(K + (size_t)(kc + sub * 16) * DHEAD + kk * 32,
                                        DHEAD, lane);
                c = __builtin_amdgcn_wmma_f32_16x16x32_bf16(
                    false, qa[kk], false, kf, (short)0, c, false, false);
            }
            s[sub] = c;
        }

        // online softmax (row = q0 + hl*8 + r, key col = kc + sub*16 + r16)
        float pl0[8], pl1[8], fac[8];
#pragma unroll
        for (int r = 0; r < 8; ++r) {
            int qrow = q0 + hl * 8 + r;
            int key0 = kc + r16;
            int key1 = kc + 16 + r16;
            float v0 = (key0 <= qrow) ? s[0][r] * 0.125f : IGNORE_VAL;
            float v1 = (key1 <= qrow) ? s[1][r] * 0.125f : IGNORE_VAL;
            float rm = fmaxf(v0, v1);
#pragma unroll
            for (int off = 1; off < 16; off <<= 1)
                rm = fmaxf(rm, __shfl_xor(rm, off, 32));
            float mnew = fmaxf(m[r], rm);
            float f = __expf(m[r] - mnew);
            float p0 = __expf(v0 - mnew);
            float p1 = __expf(v1 - mnew);
            float rs = p0 + p1;
#pragma unroll
            for (int off = 1; off < 16; off <<= 1)
                rs += __shfl_xor(rs, off, 32);
            lsum[r] = lsum[r] * f + rs;
            m[r] = mnew;
            fac[r] = f;
            pl0[r] = p0;
            pl1[r] = p1;
        }

        // rescale accumulator
#pragma unroll
        for (int nt = 0; nt < 4; ++nt)
#pragma unroll
            for (int r = 0; r < 8; ++r) o[nt][r] *= fac[r];

        // C-layout P -> row-major 16x32 bf16 tile in this wave's LDS slice
#pragma unroll
        for (int r = 0; r < 8; ++r) {
            pbuf[wv][(hl * 8 + r) * 32 + r16]      = (__bf16)pl0[r];
            pbuf[wv][(hl * 8 + r) * 32 + 16 + r16] = (__bf16)pl1[r];
        }
        asm volatile("" ::: "memory");  // keep DS store->load order (DS is in-order per wave)

        // reload P in A-fragment layout
        BF16x16 pa;
        {
            const __bf16* lp = &pbuf[wv][(size_t)r16 * 32 + hl * 8];
            pa.h[0] = *(const v8bf*)(lp);
            pa.h[1] = *(const v8bf*)(lp + 16);
        }
        asm volatile("" ::: "memory");

        // o += P(16x32) x V(32x64); B fragments contiguous from transposed V
#pragma unroll
        for (int nt = 0; nt < 4; ++nt) {
            v16bf vf = load_bt_bf16(V + (size_t)(nt * 16) * SS + kc, SS, lane);
            o[nt] = __builtin_amdgcn_wmma_f32_16x16x32_bf16(
                false, pa.v, false, vf, (short)0, o[nt], false, false);
        }
    }

    // normalize and store Z (B, NH, SS, DHEAD) in bf16
#pragma unroll
    for (int nt = 0; nt < 4; ++nt) {
        int col = nt * 16 + r16;
#pragma unroll
        for (int r = 0; r < 8; ++r) {
            int row = q0 + hl * 8 + r;
            float val = o[nt][r] / lsum[r];
            Zb[(((size_t)b * NH + h) * SS + row) * DHEAD + col] = (__bf16)val;
        }
    }
}

// ---- Kernel 3: output projection ----
// grid: (SS/64, NH, NB), block 128. Wave: 16 queries x 768 cols (48 n-tiles x K=64).
__global__ void oproj(const __bf16* __restrict__ Zb, const __bf16* __restrict__ Wot,
                      const float* __restrict__ bO, float* __restrict__ out) {
    int wv = threadIdx.x >> 5;
    int lane = threadIdx.x & 31;
    int r16 = lane & 15, hl = lane >> 4;
    int b = blockIdx.z, h = blockIdx.y;
    int q0 = (blockIdx.x * 4 + wv) * 16;

    const __bf16* Z = Zb + (((size_t)b * NH + h) * SS + q0) * DHEAD;
    const __bf16* W = Wot + (size_t)h * DMODEL * DHEAD;

    v16bf za[2];
#pragma unroll
    for (int kk = 0; kk < 2; ++kk) za[kk] = load_a_bf16(Z + kk * 32, DHEAD, lane);

    for (int nt = 0; nt < DMODEL / 16; ++nt) {
        v8f acc = zero8();
#pragma unroll
        for (int kk = 0; kk < 2; ++kk) {
            v16bf wf = load_bt_bf16(W + (size_t)(nt * 16) * DHEAD + kk * 32, DHEAD, lane);
            acc = __builtin_amdgcn_wmma_f32_16x16x32_bf16(
                false, za[kk], false, wf, (short)0, acc, false, false);
        }
        int col = nt * 16 + r16;
        float bv = bO[col] * (1.0f / (float)NH);
#pragma unroll
        for (int r = 0; r < 8; ++r) {
            int row = q0 + hl * 8 + r;
            out[(((size_t)b * SS + row) * NH + h) * DMODEL + col] = acc[r] + bv;
        }
    }
}

extern "C" void kernel_launch(void* const* d_in, const int* in_sizes, int n_in,
                              void* d_out, int out_size, void* d_ws, size_t ws_size,
                              hipStream_t stream) {
    const float* Xq = (const float*)d_in[0];
    const float* Xk = (const float*)d_in[1];
    const float* Xv = (const float*)d_in[2];
    const float* Wq = (const float*)d_in[3];
    const float* bQ = (const float*)d_in[4];
    const float* Wk = (const float*)d_in[5];
    const float* bK = (const float*)d_in[6];
    const float* Wv = (const float*)d_in[7];
    const float* bV = (const float*)d_in[8];
    const float* Wo = (const float*)d_in[9];
    const float* bO = (const float*)d_in[10];
    float* out = (float*)d_out;

    // workspace layout (bf16 elements)
    const size_t nW = (size_t)NH * DMODEL * DHEAD;        // 589,824
    const size_t nQ = (size_t)NB * NH * SS * DHEAD;       // 3,145,728
    __bf16* Wt  = (__bf16*)d_ws;      // 3*nW
    __bf16* Wot = Wt + 3 * nW;        // nW
    __bf16* Qb  = Wot + nW;           // nQ
    __bf16* Kb  = Qb + nQ;            // nQ
    __bf16* Vt  = Kb + nQ;            // nQ (transposed)
    __bf16* Zb  = Vt + nQ;            // nQ
    // total ~29.9 MB

    {
        int total = 4 * (int)nW;
        prep_weights<<<(total + 255) / 256, 256, 0, stream>>>(Wq, Wk, Wv, Wo, Wt, Wot);
    }
    qkv_proj<<<dim3(SS / 64, 3 * NH, NB), 128, 0, stream>>>(Xq, Xk, Xv, Wt, bQ, bK, bV,
                                                            Qb, Kb, Vt);
    attn<<<dim3(SS / 64, NH, NB), 128, 0, stream>>>(Qb, Kb, Vt, Zb);
    oproj<<<dim3(SS / 64, NH, NB), 128, 0, stream>>>(Zb, Wot, bO, out);
}
